// CNN_58222576664743
// MI455X (gfx1250) — compile-verified, matrix operand order checked by reading, not run on hardware
//
#include <hip/hip_runtime.h>

typedef __attribute__((ext_vector_type(16))) _Float16 v16h;
typedef __attribute__((ext_vector_type(8)))  _Float16 v8h;
typedef __attribute__((ext_vector_type(8)))  float    v8f;

#define N_NODES 8192
#define N_EDGES 65536
#define NBASIS  10
#define MID     32
#define HS      32
#define HV      8
#define INS     64
#define OS      16
#define HDIM    56
#define W0N     2560
#define W1N     1664
#define W2N     640
#define PADC    16

// ---- constants (all e3nn normalizations folded) ----
#define ISQ10  0.31622776601f   // 1/sqrt(NB)
#define ISQ32  0.17677669529f   // 1/sqrt(32)
#define ISQ8   0.35355339059f   // 1/sqrt(8)
#define ISQ3   0.57735026919f
#define ISQ2   0.70710678119f
#define GS0    0.02209708691f   // (1/sqrt(MID)) * (1/sqrt(IN_S=64))
#define GS1    0.125f           // (1/sqrt(MID)) * (1/sqrt(2))
#define GV1    0.10206207262f   // (1/sqrt(MID)) * (1/sqrt(3))
#define GS2    0.125f           // (1/sqrt(MID)) * (1/sqrt(2))
#define LNSC   0.35355339059f   // 1/sqrt(avg_neighbors = 8)

__device__ __forceinline__ v8f wmma_f16(v16h a, v16h b, v8f c) {
  return __builtin_amdgcn_wmma_f32_16x16x32_f16(false, a, false, b, (short)0, c,
                                                false, false);
}

__device__ __forceinline__ v16h scale_a(v16h a, float s) {
  _Float16 h = (_Float16)s;
  v16h r;
#pragma unroll
  for (int i = 0; i < 16; ++i) r[i] = a[i] * h;
  return r;
}

// ---------------- W2 -> fp16 B-fragment repack ----------------
// Bfrag[(c*2 + laneHalf)*16 + i] = W2[k(i,half)*WN + c], k = half*8 + (i<8?i:i+8)
__global__ void k_prep(const float* __restrict__ W2, _Float16* __restrict__ Bf, int WN) {
  int idx = blockIdx.x * blockDim.x + threadIdx.x;
  int total = (WN + PADC) * 2;
  if (idx >= total) return;
  int c = idx >> 1, hf = idx & 1;
  _Float16* o = Bf + (size_t)(c * 2 + hf) * 16;
#pragma unroll
  for (int i = 0; i < 16; ++i) {
    int k = hf * 8 + (i < 8 ? i : i + 8);
    float v = (c < WN) ? W2[(size_t)k * WN + c] : 0.0f;
    o[i] = (_Float16)v;
  }
}

// ---------------- node embedding: concat(Ez[z], Em[mol]) -> fp16 ----------------
__global__ void k_nodes(const int* __restrict__ z, const int* __restrict__ mol,
                        const float* __restrict__ Ez, const float* __restrict__ Em,
                        _Float16* __restrict__ x0h) {
  int n = blockIdx.x * blockDim.x + threadIdx.x;
  if (n >= N_NODES) return;
  int zz = z[n], mm = mol[n];
#pragma unroll
  for (int j = 0; j < 48; ++j) x0h[n * 64 + j] = (_Float16)Ez[zz * 48 + j];
#pragma unroll
  for (int j = 0; j < 16; ++j) x0h[n * 64 + 48 + j] = (_Float16)Em[mm * 16 + j];
}

// ---------------- per-edge geometry + radial MLP layer-1 ----------------
__global__ void k_geom(const float* __restrict__ pos, const int* __restrict__ esrc,
                       const int* __restrict__ edst,
                       const float* __restrict__ W1a, const float* __restrict__ W1b,
                       const float* __restrict__ W1c,
                       float* __restrict__ sh, _Float16* __restrict__ h0,
                       _Float16* __restrict__ h1, _Float16* __restrict__ h2) {
  int e = blockIdx.x * blockDim.x + threadIdx.x;
  if (e >= N_EDGES) return;
  int s = esrc[e], d = edst[e];
  float vx = pos[d * 3 + 0] - pos[s * 3 + 0];
  float vy = pos[d * 3 + 1] - pos[s * 3 + 1];
  float vz = pos[d * 3 + 2] - pos[s * 3 + 2];
  float len = sqrtf(vx * vx + vy * vy + vz * vz);
  float inv = 1.0f / fmaxf(len, 1e-9f);
  const float SQ3 = 1.7320508f;
  sh[e * 3 + 0] = SQ3 * vx * inv;
  sh[e * 3 + 1] = SQ3 * vy * inv;
  sh[e * 3 + 2] = SQ3 * vz * inv;
  // smooth_finite soft-one-hot * sqrt(NB)
  float emb[NBASIS];
  const float STEP = 10.0f / 11.0f;
  const float C = 1.14136f * 7.38905610f * 3.16227766f;  // 1.14136*e^2*sqrt(10)
#pragma unroll
  for (int j = 0; j < NBASIS; ++j) {
    float val = STEP * (float)(j + 1);
    float u = (len - val) / STEP;
    float ap = u + 1.0f, bp = 1.0f - u;
    float fa = (ap > 0.f) ? __expf(-1.0f / ap) : 0.f;
    float fb = (bp > 0.f) ? __expf(-1.0f / bp) : 0.f;
    emb[j] = C * fa * fb;
  }
  const float* W1s[3] = {W1a, W1b, W1c};
  _Float16* outs[3] = {h0, h1, h2};
#pragma unroll
  for (int blk = 0; blk < 3; ++blk) {
    _Float16* o = outs[blk] + (size_t)e * 32;
#pragma unroll
    for (int m = 0; m < MID; ++m) {
      float acc = 0.f;
#pragma unroll
      for (int j = 0; j < NBASIS; ++j) acc += emb[j] * W1s[blk][j * MID + m];
      acc *= ISQ10;
      float h = acc / (1.0f + __expf(-acc));  // silu
      o[m] = (_Float16)h;
    }
  }
}

// ---- stage h tiles through LDS; build A fragment with the K-half permute ----
__device__ __forceinline__ void stage_tiles(const _Float16* __restrict__ hfr,
                                            int eBase, int wv, int lane,
                                            _Float16* hsh, int nTiles) {
  const float4* gsrc = (const float4*)(hfr + (size_t)eBase * 32);
  float4* ldst = (float4*)(hsh + wv * 512 * nTiles);
  for (int t = lane; t < 64 * nTiles; t += 32) ldst[t] = gsrc[t];
  __syncthreads();
}

__device__ __forceinline__ v16h read_afrag(const _Float16* hw, int tile, int lane) {
  int row = lane & 15, hf = lane >> 4;
  const _Float16* base = hw + tile * 512 + row * 32 + hf * 8;
  v8h alo = *(const v8h*)(base);       // K = hf*8 .. hf*8+7
  v8h ahi = *(const v8h*)(base + 16);  // K = hf*8+16 .. hf*8+23
  v16h a;
#pragma unroll
  for (int i = 0; i < 8; ++i) { a[i] = alo[i]; a[i + 8] = ahi[i]; }
  return a;
}

// ---------------- block 0 messages: 64x0e -> 32x0e + 8x1o ----------------
// 2 edge tiles per wave share each B fragment; B double-buffered in registers.
__global__ void __launch_bounds__(256) k_msg0(const _Float16* __restrict__ hfr,
    const _Float16* __restrict__ Bf, const _Float16* __restrict__ x0h,
    const int* __restrict__ esrc, const int* __restrict__ edst,
    const float* __restrict__ sh, float* __restrict__ acc1) {
  __shared__ __align__(16) _Float16 hsh[8 * 1024];
  int lane = threadIdx.x & 31;
  int wv = threadIdx.x >> 5;
  int e0 = (blockIdx.x * 8 + wv) * 32;
  int row = lane & 15, hf = lane >> 4;
  stage_tiles(hfr, e0, wv, lane, hsh, 2);
  const _Float16* hw = hsh + wv * 1024;
  v16h a0 = read_afrag(hw, 0, lane);
  v16h a1 = read_afrag(hw, 1, lane);
  const _Float16* xr0 = x0h + (size_t)esrc[e0 + row] * 64;
  const _Float16* xr1 = x0h + (size_t)esrc[e0 + 16 + row] * 64;
  const v16h* B = (const v16h*)Bf;
  size_t ci = (size_t)row * 2 + hf;
  v16h b0n = B[ci];            // c = u*32 + row       (stride 64 / u)
  v16h b1n = B[ci + 32];       // c = u*32 + 16 + row
  v16h b2n = B[ci + 4096];     // c = 2048 + u*8 + row (stride 16 / u)
  v8f s00 = {}, s01 = {}, s02 = {}, s10 = {}, s11 = {}, s12 = {};
  for (int u = 0; u < INS; ++u) {
    v16h b0 = b0n, b1 = b1n, b2 = b2n;
    int un = (u < INS - 1) ? u + 1 : u;
    size_t bs = ci + (size_t)un * 64;
    b0n = B[bs];
    b1n = B[bs + 32];
    b2n = B[ci + 4096 + (size_t)un * 16];
    __builtin_prefetch(&B[ci + (size_t)(un + 1 < INS ? un + 1 : un) * 64], 0, 1);
    v16h as0 = scale_a(a0, (float)xr0[u]);
    s00 = wmma_f16(as0, b0, s00);
    s01 = wmma_f16(as0, b1, s01);
    s02 = wmma_f16(as0, b2, s02);
    v16h as1 = scale_a(a1, (float)xr1[u]);
    s10 = wmma_f16(as1, b0, s10);
    s11 = wmma_f16(as1, b1, s11);
    s12 = wmma_f16(as1, b2, s12);
  }
  int nn = lane & 15;
#pragma unroll
  for (int r = 0; r < 8; ++r) {
    {
      int em = e0 + r + 8 * hf;
      int dn = edst[em];
      float* na = acc1 + (size_t)dn * HDIM;
      atomicAdd(na + nn, s00[r] * GS0);
      atomicAdd(na + 16 + nn, s01[r] * GS0);
      if (nn < 8) {
        float t = s02[r] * GS0;
        atomicAdd(na + 32 + nn * 3 + 0, t * sh[em * 3 + 0]);
        atomicAdd(na + 32 + nn * 3 + 1, t * sh[em * 3 + 1]);
        atomicAdd(na + 32 + nn * 3 + 2, t * sh[em * 3 + 2]);
      }
    }
    {
      int em = e0 + 16 + r + 8 * hf;
      int dn = edst[em];
      float* na = acc1 + (size_t)dn * HDIM;
      atomicAdd(na + nn, s10[r] * GS0);
      atomicAdd(na + 16 + nn, s11[r] * GS0);
      if (nn < 8) {
        float t = s12[r] * GS0;
        atomicAdd(na + 32 + nn * 3 + 0, t * sh[em * 3 + 0]);
        atomicAdd(na + 32 + nn * 3 + 1, t * sh[em * 3 + 1]);
        atomicAdd(na + 32 + nn * 3 + 2, t * sh[em * 3 + 2]);
      }
    }
  }
}

// ---------------- block 1 messages: 32x0e+8x1o -> 32x0e+8x1o ----------------
__global__ void __launch_bounds__(256) k_msg1(const _Float16* __restrict__ hfr,
    const _Float16* __restrict__ Bf, const float* __restrict__ x1,
    const int* __restrict__ esrc, const int* __restrict__ edst,
    const float* __restrict__ sh, float* __restrict__ acc2) {
  __shared__ __align__(16) _Float16 hsh[8 * 512];
  int lane = threadIdx.x & 31;
  int wv = threadIdx.x >> 5;
  int e0 = (blockIdx.x * 8 + wv) * 16;
  int row = lane & 15, hf = lane >> 4;
  stage_tiles(hfr, e0, wv, lane, hsh, 1);
  v16h a = read_afrag(hsh + wv * 512, 0, lane);
  int ea = e0 + row;
  const float* xr = x1 + (size_t)esrc[ea] * HDIM;
  const v16h* B = (const v16h*)Bf;
  float sx = sh[ea * 3 + 0], sy = sh[ea * 3 + 1], sz = sh[ea * 3 + 2];
  size_t ci = (size_t)row * 2 + hf;
  v8f s0 = {}, s1 = {}, t1 = {};
  v8f tvv[3] = {{}, {}, {}};
  // scalar features xs (w_ss) + sv path (w_sv), B double-buffered
  v16h b0n = B[ci];             // c = u*32 + row
  v16h b1n = B[ci + 32];        // c = u*32 + 16 + row
  v16h btn = B[ci + 2560];      // c = 1280 + u*8 + row
  for (int u = 0; u < HS; ++u) {
    v16h b0 = b0n, b1 = b1n, bt = btn;
    int un = (u < HS - 1) ? u + 1 : u;
    size_t bs = ci + (size_t)un * 64;
    b0n = B[bs];
    b1n = B[bs + 32];
    btn = B[ci + 2560 + (size_t)un * 16];
    __builtin_prefetch(&B[ci + 2048 + (size_t)(u & 7) * 64], 0, 1);
    v16h as = scale_a(a, xr[u] * ISQ32);
    s0 = wmma_f16(as, b0, s0);
    s1 = wmma_f16(as, b1, s1);
    t1 = wmma_f16(as, bt, t1);
  }
  // CG products for vector features of this lane's A-row edge
  float xv[HV][3], dot[HV], crs[HV][3];
#pragma unroll
  for (int u = 0; u < HV; ++u) {
    xv[u][0] = xr[32 + u * 3 + 0];
    xv[u][1] = xr[32 + u * 3 + 1];
    xv[u][2] = xr[32 + u * 3 + 2];
    dot[u] = (xv[u][0] * sx + xv[u][1] * sy + xv[u][2] * sz) * ISQ3;
    crs[u][0] = (xv[u][1] * sz - xv[u][2] * sy) * ISQ2;
    crs[u][1] = (xv[u][2] * sx - xv[u][0] * sz) * ISQ2;
    crs[u][2] = (xv[u][0] * sy - xv[u][1] * sx) * ISQ2;
  }
  // dot features (w_vvs): c = 1024 + u*32 + {row, 16+row}  (unrolled; scheduler hoists loads)
#pragma unroll
  for (int u = 0; u < HV; ++u) {
    v16h as = scale_a(a, dot[u] * ISQ8);
    size_t bs = ci + 2048 + (size_t)u * 64;
    s0 = wmma_f16(as, B[bs], s0);
    s1 = wmma_f16(as, B[bs + 32], s1);
  }
  // xv (w_vs, c=1536+u*8) + crs (w_vvv, c=1600+u*8) per vector component
#pragma unroll
  for (int u = 0; u < HV; ++u) {
#pragma unroll
    for (int i = 0; i < 3; ++i) {
      v16h a1 = scale_a(a, xv[u][i] * ISQ8);
      tvv[i] = wmma_f16(a1, B[ci + 3072 + (size_t)u * 16], tvv[i]);
      v16h a2 = scale_a(a, crs[u][i] * ISQ8);
      tvv[i] = wmma_f16(a2, B[ci + 3200 + (size_t)u * 16], tvv[i]);
    }
  }
  int nn = lane & 15;
#pragma unroll
  for (int r = 0; r < 8; ++r) {
    int em = e0 + r + 8 * hf;
    int dn = edst[em];
    float* na = acc2 + (size_t)dn * HDIM;
    atomicAdd(na + nn, s0[r] * GS1);
    atomicAdd(na + 16 + nn, s1[r] * GS1);
    if (nn < 8) {
      float te = t1[r];
      float ex = sh[em * 3 + 0], ey = sh[em * 3 + 1], ez = sh[em * 3 + 2];
      atomicAdd(na + 32 + nn * 3 + 0, (te * ex + tvv[0][r]) * GV1);
      atomicAdd(na + 32 + nn * 3 + 1, (te * ey + tvv[1][r]) * GV1);
      atomicAdd(na + 32 + nn * 3 + 2, (te * ez + tvv[2][r]) * GV1);
    }
  }
}

// ---------------- block 2 messages: 32x0e+8x1o -> 16x0e ----------------
// 2 edge tiles per wave; single B stream double-buffered.
__global__ void __launch_bounds__(256) k_msg2(const _Float16* __restrict__ hfr,
    const _Float16* __restrict__ Bf, const float* __restrict__ x2,
    const int* __restrict__ esrc, const int* __restrict__ edst,
    const float* __restrict__ sh, float* __restrict__ acc3) {
  __shared__ __align__(16) _Float16 hsh[8 * 1024];
  int lane = threadIdx.x & 31;
  int wv = threadIdx.x >> 5;
  int e0 = (blockIdx.x * 8 + wv) * 32;
  int row = lane & 15, hf = lane >> 4;
  stage_tiles(hfr, e0, wv, lane, hsh, 2);
  const _Float16* hw = hsh + wv * 1024;
  v16h a0 = read_afrag(hw, 0, lane);
  v16h a1 = read_afrag(hw, 1, lane);
  int ea0 = e0 + row, ea1 = e0 + 16 + row;
  const float* xr0 = x2 + (size_t)esrc[ea0] * HDIM;
  const float* xr1 = x2 + (size_t)esrc[ea1] * HDIM;
  const v16h* B = (const v16h*)Bf;
  size_t ci = (size_t)row * 2 + hf;
  v8f p0 = {}, p1 = {};
  v16h bn = B[ci];  // c = u*16 + row (stride 32 / u)
  for (int u = 0; u < HS; ++u) {
    v16h b = bn;
    int un = (u < HS - 1) ? u + 1 : u;
    bn = B[ci + (size_t)un * 32];
    __builtin_prefetch(&B[ci + 1024 + (size_t)(u & 7) * 32], 0, 1);
    p0 = wmma_f16(scale_a(a0, xr0[u] * ISQ32), b, p0);
    p1 = wmma_f16(scale_a(a1, xr1[u] * ISQ32), b, p1);
  }
  float sx0 = sh[ea0 * 3 + 0], sy0 = sh[ea0 * 3 + 1], sz0 = sh[ea0 * 3 + 2];
  float sx1 = sh[ea1 * 3 + 0], sy1 = sh[ea1 * 3 + 1], sz1 = sh[ea1 * 3 + 2];
#pragma unroll
  for (int u = 0; u < HV; ++u) {
    v16h b = B[ci + 1024 + (size_t)u * 32];  // c = 512 + u*16 + row
    float d0 = (xr0[32 + u * 3 + 0] * sx0 + xr0[32 + u * 3 + 1] * sy0 +
                xr0[32 + u * 3 + 2] * sz0) * ISQ3;
    p0 = wmma_f16(scale_a(a0, d0 * ISQ8), b, p0);
    float d1 = (xr1[32 + u * 3 + 0] * sx1 + xr1[32 + u * 3 + 1] * sy1 +
                xr1[32 + u * 3 + 2] * sz1) * ISQ3;
    p1 = wmma_f16(scale_a(a1, d1 * ISQ8), b, p1);
  }
  int nn = lane & 15;
#pragma unroll
  for (int r = 0; r < 8; ++r) {
    int em0 = e0 + r + 8 * hf;
    atomicAdd(acc3 + (size_t)edst[em0] * OS + nn, p0[r] * GS2);
    int em1 = e0 + 16 + r + 8 * hf;
    atomicAdd(acc3 + (size_t)edst[em1] * OS + nn, p1[r] * GS2);
  }
}

// ---------------- layernorm over last dim (with 1/sqrt(avg) fold) ----------------
__global__ void k_ln(const float* __restrict__ acc, const float* __restrict__ g,
                     const float* __restrict__ b, float* __restrict__ out, int dim) {
  int n = blockIdx.x * blockDim.x + threadIdx.x;
  if (n >= N_NODES) return;
  float s = 0.f, s2 = 0.f;
  for (int d = 0; d < dim; ++d) {
    float v = acc[(size_t)n * dim + d] * LNSC;
    s += v; s2 += v * v;
  }
  float m = s / (float)dim;
  float var = s2 / (float)dim - m * m;
  float r = rsqrtf(var + 1e-5f);
  for (int d = 0; d < dim; ++d) {
    float v = acc[(size_t)n * dim + d] * LNSC;
    out[(size_t)n * dim + d] = (v - m) * r * g[d] + b[d];
  }
}

extern "C" void kernel_launch(void* const* d_in, const int* in_sizes, int n_in,
                              void* d_out, int out_size, void* d_ws, size_t ws_size,
                              hipStream_t stream) {
  (void)in_sizes; (void)n_in; (void)out_size; (void)ws_size;
  const float* pos  = (const float*)d_in[0];
  const int*   z    = (const int*)d_in[1];
  const int*   mol  = (const int*)d_in[2];
  const int*   esrc = (const int*)d_in[3];
  const int*   edst = (const int*)d_in[4];
  const float* Ez   = (const float*)d_in[5];
  const float* Em   = (const float*)d_in[6];
  const float* b0W1 = (const float*)d_in[7];
  const float* b0W2 = (const float*)d_in[8];
  const float* b0g  = (const float*)d_in[9];
  const float* b0b  = (const float*)d_in[10];
  const float* b1W1 = (const float*)d_in[11];
  const float* b1W2 = (const float*)d_in[12];
  const float* b1g  = (const float*)d_in[13];
  const float* b1b  = (const float*)d_in[14];
  const float* b2W1 = (const float*)d_in[15];
  const float* b2W2 = (const float*)d_in[16];
  const float* b2g  = (const float*)d_in[17];
  const float* b2b  = (const float*)d_in[18];

  char* ws = (char*)d_ws;
  size_t off = 0;
  auto take = [&](size_t bytes) -> char* {
    char* p = ws + off;
    off += (bytes + 255) & ~(size_t)255;
    return p;
  };
  float*    sh   = (float*)take((size_t)N_EDGES * 3 * sizeof(float));
  _Float16* h0   = (_Float16*)take((size_t)N_EDGES * 32 * sizeof(_Float16));
  _Float16* h1   = (_Float16*)take((size_t)N_EDGES * 32 * sizeof(_Float16));
  _Float16* h2   = (_Float16*)take((size_t)N_EDGES * 32 * sizeof(_Float16));
  _Float16* x0h  = (_Float16*)take((size_t)N_NODES * 64 * sizeof(_Float16));
  _Float16* Bf0  = (_Float16*)take((size_t)(W0N + PADC) * 32 * sizeof(_Float16));
  _Float16* Bf1  = (_Float16*)take((size_t)(W1N + PADC) * 32 * sizeof(_Float16));
  _Float16* Bf2  = (_Float16*)take((size_t)(W2N + PADC) * 32 * sizeof(_Float16));
  float*    acc1 = (float*)take((size_t)N_NODES * HDIM * sizeof(float));
  float*    x1   = (float*)take((size_t)N_NODES * HDIM * sizeof(float));
  float*    acc2 = (float*)take((size_t)N_NODES * HDIM * sizeof(float));
  float*    x2   = (float*)take((size_t)N_NODES * HDIM * sizeof(float));
  float*    acc3 = (float*)take((size_t)N_NODES * OS * sizeof(float));

  hipMemsetAsync(acc1, 0, (size_t)N_NODES * HDIM * sizeof(float), stream);
  hipMemsetAsync(acc2, 0, (size_t)N_NODES * HDIM * sizeof(float), stream);
  hipMemsetAsync(acc3, 0, (size_t)N_NODES * OS * sizeof(float), stream);

  k_prep<<<((W0N + PADC) * 2 + 255) / 256, 256, 0, stream>>>(b0W2, Bf0, W0N);
  k_prep<<<((W1N + PADC) * 2 + 255) / 256, 256, 0, stream>>>(b1W2, Bf1, W1N);
  k_prep<<<((W2N + PADC) * 2 + 255) / 256, 256, 0, stream>>>(b2W2, Bf2, W2N);
  k_nodes<<<N_NODES / 256, 256, 0, stream>>>(z, mol, Ez, Em, x0h);
  k_geom<<<N_EDGES / 256, 256, 0, stream>>>(pos, esrc, edst, b0W1, b1W1, b2W1,
                                            sh, h0, h1, h2);

  const int T16 = N_EDGES / 16;  // 4096 tiles
  k_msg0<<<T16 / 16, 256, 0, stream>>>(h0, Bf0, x0h, esrc, edst, sh, acc1);
  k_ln<<<N_NODES / 256, 256, 0, stream>>>(acc1, b0g, b0b, x1, HDIM);
  k_msg1<<<T16 / 8, 256, 0, stream>>>(h1, Bf1, x1, esrc, edst, sh, acc2);
  k_ln<<<N_NODES / 256, 256, 0, stream>>>(acc2, b1g, b1b, x2, HDIM);
  k_msg2<<<T16 / 16, 256, 0, stream>>>(h2, Bf2, x2, esrc, edst, sh, acc3);
  k_ln<<<N_NODES / 256, 256, 0, stream>>>(acc3, b2g, b2b, (float*)d_out, OS);
}
